// SchNet_29618094473608
// MI455X (gfx1250) — matrix operand order repaired and checked by reading, not compiled, not run on hardware
//
#include <hip/hip_runtime.h>
#include <hip/hip_bf16.h>
#include <math.h>

#define DEVINL __device__ __forceinline__

typedef __attribute__((ext_vector_type(16))) __bf16 v16bf;
typedef __attribute__((ext_vector_type(8)))  float  v8f;

#define HIDD 128
#define NFF  32
#define NGG  50
#define NLL  6
#define CUT  10.0f
// delta = CUT/(NG-1); coeff = -0.5/delta^2
#define GDELTA (CUT / (float)(NGG - 1))
#define GCOEFF (-0.5f * (float)(NGG - 1) * (float)(NGG - 1) / (CUT * CUT))

#if __has_builtin(__builtin_amdgcn_tensor_load_to_lds)
#define HAVE_TDM 1
#else
#define HAVE_TDM 0
#endif

// ---- softplus(x) - ln(2) ----
DEVINL float sspf(float x) {
    float sp = (x > 20.0f) ? x : log1pf(__expf(x));
    return sp - 0.69314718055994531f;
}

// ---- WMMA fragment index maps (contiguous k order used by this file, 16-bit operands) ----
// A 16x32 (MxK): lane m = lane&15, kg = lane>>4.
// element i -> K = i + 16*kg
DEVINL int ka_map(int i, int kg) { return i + 16 * kg; }
// B 32x16 (KxN): lane n = lane&15, kg = lane>>4. element i -> K = i + 16*kg
DEVINL int kb_map(int i, int kg) { return i + 16 * kg; }

// A fragment from row-major f32 (K multiple of 32, no bounds check, float2 pairs)
DEVINL v16bf load_a_f32(const float* __restrict__ A, int row, int ldk,
                        int k0, int kg) {
    const float* p = A + (size_t)row * ldk + k0;
    v16bf a;
#pragma unroll
    for (int q = 0; q < 8; ++q) {
        int k = ka_map(2 * q, kg);          // pairs are K-contiguous
        float2 v = *(const float2*)(p + k);
        a[2 * q]     = (__bf16)v.x;
        a[2 * q + 1] = (__bf16)v.y;
    }
    return a;
}

// B fragment from row-major f32 B[K][ldn] (K multiple of 32)
DEVINL v16bf load_b_f32(const float* __restrict__ B, int col, int ldn,
                        int k0, int kg) {
    v16bf b;
#pragma unroll
    for (int i = 0; i < 16; ++i) {
        int k = k0 + kb_map(i, kg);
        b[i] = (__bf16)B[(size_t)k * ldn + col];
    }
    return b;
}

DEVINL v8f wmma_bf16(v16bf a, v16bf b, v8f c) {
    return __builtin_amdgcn_wmma_f32_16x16x32_bf16(false, a, false, b,
                                                   (short)0, c, false, false);
}

// ---- Tensor Data Mover: 2D f32 tile (whole matrix) global -> LDS ----
// D# per CDNA5 ISA 8.3/8.4: group0 {count=1, lds_addr, global_addr, type=2},
// group1 {data_size=4B, tensor_dim0/1, tile_dim0/1, dim0 stride}.
DEVINL void tdm_load_2d(unsigned lds_off, const float* gsrc,
                        int rowlen, int nrows, int rowstride) {
#if HAVE_TDM
    typedef __attribute__((ext_vector_type(4))) unsigned int uv4;
    typedef __attribute__((ext_vector_type(4))) int iv4;
    typedef __attribute__((ext_vector_type(8))) int iv8;
    unsigned long long ga = (unsigned long long)(uintptr_t)gsrc;
    uv4 g0;
    g0[0] = 1u;                                        // count=1 (user mode)
    g0[1] = lds_off;                                   // lds_addr (bytes)
    g0[2] = (unsigned)ga;                              // global_addr[31:0]
    g0[3] = (unsigned)((ga >> 32) & 0x01FFFFFFull)     // global_addr[56:32]
          | 0x80000000u;                               // type=2 ("image")
    iv8 g1;
    g1[0] = 0x20000;                                   // data_size=2 (4B)
    g1[1] = (rowlen & 0xffff) << 16;                   // tensor_dim0[15:0]
    g1[2] = (nrows & 0xffff) << 16;                    // tensor_dim1[15:0]
    g1[3] = (rowlen & 0xffff) << 16;                   // tile_dim0
    g1[4] = (nrows & 0xffff);                          // tile_dim1
    g1[5] = rowstride;                                 // tensor_dim0_stride lo
    g1[6] = 0;
    g1[7] = 0;
    iv4 z4 = {0, 0, 0, 0};
#if defined(__clang_major__) && (__clang_major__ >= 23)
    iv8 z8 = {0, 0, 0, 0, 0, 0, 0, 0};
    __builtin_amdgcn_tensor_load_to_lds(g0, g1, z4, z4, z8, 0);
#else
    __builtin_amdgcn_tensor_load_to_lds(g0, g1, z4, z4, 0);
#endif
#endif
}

// =====================  elementwise kernels =====================
__global__ void init_h_kernel(const int* __restrict__ z,
                              const float* __restrict__ emb,
                              float* __restrict__ h, int total) {
    int i = blockIdx.x * blockDim.x + threadIdx.x;
    if (i >= total) return;
    int node = i / HIDD, f = i % HIDD;
    h[i] = emb[z[node] * HIDD + f];
}

__global__ void edge_geom_kernel(const float* __restrict__ pos,
                                 const int* __restrict__ src,
                                 const int* __restrict__ dst,
                                 const float* __restrict__ mask,
                                 float* __restrict__ dArr,
                                 float* __restrict__ Carr, int E) {
    int e = blockIdx.x * blockDim.x + threadIdx.x;
    if (e >= E) return;
    int s = src[e], t = dst[e];
    float dx = pos[s * 3 + 0] - pos[t * 3 + 0];
    float dy = pos[s * 3 + 1] - pos[t * 3 + 1];
    float dz = pos[s * 3 + 2] - pos[t * 3 + 2];
    float d = sqrtf(dx * dx + dy * dy + dz * dz);
    dArr[e] = d;
    Carr[e] = 0.5f * (cosf(d * (float)M_PI / CUT) + 1.0f) * mask[e];
}

__global__ void zero_kernel(float* __restrict__ p, int total) {
    int i = blockIdx.x * blockDim.x + threadIdx.x;
    if (i < total) p[i] = 0.0f;
}

// =====================  WMMA GEMM (one wave per 16x16 tile) =====================
// EPI 0: Out = A@B ; EPI 1: Out = ssp(A@B+bias) ; EPI 2: Out += A@B + bias
template <int EPI>
__global__ void gemm_wmma_kernel(const float* __restrict__ A,
                                 const float* __restrict__ B,
                                 const float* __restrict__ bias,
                                 float* __restrict__ Out,
                                 int M, int K, int Nc,
                                 int totalTiles, int tilesN) {
    int lane = threadIdx.x & 31;
    int wid  = blockIdx.x * (blockDim.x >> 5) + (threadIdx.x >> 5);
    if (wid >= totalTiles) return;           // wave-uniform exit
    int tm = wid / tilesN, tn = wid % tilesN;
    int m16 = lane & 15, kg = lane >> 4;
    int row = tm * 16 + m16;
    int col = tn * 16 + m16;

    v8f acc = {};
    for (int k0 = 0; k0 < K; k0 += 32) {
        v16bf a = load_a_f32(A, row, K, k0, kg);
        v16bf b = load_b_f32(B, col, Nc, k0, kg);
        acc = wmma_bf16(a, b, acc);
    }
    float bv = (EPI != 0) ? bias[col] : 0.0f;
#pragma unroll
    for (int r = 0; r < 8; ++r) {
        int orow = tm * 16 + r + 8 * kg;
        float v = acc[r] + bv;
        if (EPI == 1) v = sspf(v);
        if (EPI == 2) v += Out[orow * Nc + col];
        Out[orow * Nc + col] = v;
    }
}

// =====================  fused edge kernel =====================
// Per 16-edge tile (one wave):
//   ea(16x64 gaussian, built in regs) @ w1 + b1 -> ssp -> LDS transpose
//   -> @ w2 + b2 -> *C -> msg = hs[src]*W -> atomic add into m[dst]
// Weights staged via TDM into LDS, pre-swizzled to per-lane bf16 fragments.
// Per-edge scalars (C, src, dst) kept in registers; epilogue rows fetched
// via cross-lane shuffles (no LDS staging -> no address spills).
#define EWAVES 4
__global__ void edge_kernel(const float* __restrict__ dArr,
                            const float* __restrict__ Carr,
                            const int* __restrict__ src,
                            const int* __restrict__ dst,
                            const float* __restrict__ w1,
                            const float* __restrict__ b1,
                            const float* __restrict__ w2,
                            const float* __restrict__ b2,
                            const float* __restrict__ hs,
                            float* __restrict__ mOut,
                            int nTiles) {
    __shared__ float ldsW1raw[NGG * NFF];          // 6400 B (TDM dest)
    __shared__ float ldsW2raw[NFF * NFF];          // 4096 B (TDM dest)
    __shared__ v16bf fragW1[2][2][32];             // [kstep][ntile][lane] 4 KB
    __shared__ v16bf fragW2[2][32];                // [ntile][lane]        2 KB
    __shared__ float ldsT[EWAVES][16][NFF + 1];    // transpose buffer

    int lane = threadIdx.x & 31;
    int wv   = threadIdx.x >> 5;
    int wid0 = blockIdx.x * EWAVES + wv;
    int nWaves = gridDim.x * EWAVES;
    int m16 = lane & 15, kg = lane >> 4;

    // ---- stage raw weights into LDS (wave 0) ----
    if (wv == 0) {
#if HAVE_TDM
        tdm_load_2d((unsigned)(uintptr_t)(void*)ldsW1raw, w1, NFF, NGG, NFF);
        tdm_load_2d((unsigned)(uintptr_t)(void*)ldsW2raw, w2, NFF, NFF, NFF);
        __builtin_amdgcn_s_wait_tensorcnt(0);
#else
        for (int i = lane; i < NGG * NFF; i += 32) ldsW1raw[i] = w1[i];
        for (int i = lane; i < NFF * NFF; i += 32) ldsW2raw[i] = w2[i];
#endif
    }
    __syncthreads();

    // ---- pre-swizzle per-lane B fragments (wave 0; identical for all waves) ----
    if (wv == 0) {
#pragma unroll
        for (int s = 0; s < 2; ++s)
#pragma unroll
            for (int n = 0; n < 2; ++n) {
                v16bf f;
#pragma unroll
                for (int i = 0; i < 16; ++i) {
                    int k = s * 32 + 16 * kg + i;
                    f[i] = (__bf16)((k < NGG) ? ldsW1raw[k * NFF + n * 16 + m16]
                                              : 0.0f);
                }
                fragW1[s][n][lane] = f;
            }
#pragma unroll
        for (int n = 0; n < 2; ++n) {
            v16bf f;
#pragma unroll
            for (int i = 0; i < 16; ++i) {
                int k = 16 * kg + i;
                f[i] = (__bf16)ldsW2raw[k * NFF + n * 16 + m16];
            }
            fragW2[n][lane] = f;
        }
    }
    __syncthreads();

    float b1v0 = b1[m16], b1v1 = b1[16 + m16];
    float b2v0 = b2[m16], b2v1 = b2[16 + m16];

    for (int t = wid0; t < nTiles; t += nWaves) {
        int e0 = t * 16;
        // per-edge scalars live in registers: lane holds edge e0 + (lane&15)
        float Cv = Carr[e0 + m16];
        int   Sv = src[e0 + m16];
        int   Dv = dst[e0 + m16];
        float de = dArr[e0 + m16];

        // A fragments: ea[k] = exp(coeff*(d - k*delta)^2), K padded 50->64
        v16bf a0, a1;
#pragma unroll
        for (int i = 0; i < 16; ++i) {
            int k = ka_map(i, kg);
            float t0 = de - (float)k * GDELTA;
            a0[i] = (__bf16)((k < NGG) ? __expf(GCOEFF * t0 * t0) : 0.0f);
            int k2 = k + 32;
            float t1 = de - (float)k2 * GDELTA;
            a1[i] = (__bf16)((k2 < NGG) ? __expf(GCOEFF * t1 * t1) : 0.0f);
        }

        // first edge GEMM: t = ea @ w1 + b1 (accumulators seeded with bias);
        // B fragments pulled from LDS just-in-time to keep VGPR pressure low.
        v8f c0, c1;
#pragma unroll
        for (int i = 0; i < 8; ++i) { c0[i] = b1v0; c1[i] = b1v1; }
        c0 = wmma_bf16(a0, fragW1[0][0][lane], c0);
        c0 = wmma_bf16(a1, fragW1[1][0][lane], c0);
        c1 = wmma_bf16(a0, fragW1[0][1][lane], c1);
        c1 = wmma_bf16(a1, fragW1[1][1][lane], c1);

        // ssp + transpose through LDS (D layout -> A layout)
#pragma unroll
        for (int r = 0; r < 8; ++r) {
            int er = r + 8 * kg;
            ldsT[wv][er][m16]      = sspf(c0[r]);
            ldsT[wv][er][16 + m16] = sspf(c1[r]);
        }
        // intra-wave LDS RAW is safe: DS ops are in-order within a wave

        v16bf at;
#pragma unroll
        for (int i = 0; i < 16; ++i)
            at[i] = (__bf16)ldsT[wv][m16][ka_map(i, kg)];

        // second edge GEMM: W = t @ w2 + b2
        v8f d0, d1;
#pragma unroll
        for (int i = 0; i < 8; ++i) { d0[i] = b2v0; d1[i] = b2v1; }
        d0 = wmma_bf16(at, fragW2[0][lane], d0);
        d1 = wmma_bf16(at, fragW2[1][lane], d1);

        // epilogue: *C, gather hs[src], atomic scatter into m[dst].
        // Row er = r + 8*kg scalars come from lane er via shuffle (no LDS).
#pragma unroll
        for (int r = 0; r < 8; ++r) {
            int srcLane = r + 8 * kg;
            float Ce = __shfl(Cv, srcLane, 32);
            int   sN = __shfl(Sv, srcLane, 32);
            int   dN = __shfl(Dv, srcLane, 32);
            float W0 = d0[r] * Ce;
            float W1 = d1[r] * Ce;
            float msg0 = hs[sN * NFF + m16] * W0;
            float msg1 = hs[sN * NFF + 16 + m16] * W1;
            unsafeAtomicAdd(&mOut[dN * NFF + m16], msg0);
            unsafeAtomicAdd(&mOut[dN * NFF + 16 + m16], msg1);
        }
    }
}

// =====================  host launcher =====================
static inline void launch_gemm(int epi, const float* A, const float* B,
                               const float* bias, float* Out,
                               int M, int K, int Nc, hipStream_t stream) {
    int tilesM = (M + 15) / 16, tilesN = (Nc + 15) / 16;
    int totalTiles = tilesM * tilesN;
    int blocks = (totalTiles + 3) / 4;  // 4 waves (128 threads) per block
    switch (epi) {
    case 0:
        gemm_wmma_kernel<0><<<blocks, 128, 0, stream>>>(A, B, bias, Out, M, K, Nc, totalTiles, tilesN);
        break;
    case 1:
        gemm_wmma_kernel<1><<<blocks, 128, 0, stream>>>(A, B, bias, Out, M, K, Nc, totalTiles, tilesN);
        break;
    default:
        gemm_wmma_kernel<2><<<blocks, 128, 0, stream>>>(A, B, bias, Out, M, K, Nc, totalTiles, tilesN);
        break;
    }
}

extern "C" void kernel_launch(void* const* d_in, const int* in_sizes, int n_in,
                              void* d_out, int out_size, void* d_ws, size_t ws_size,
                              hipStream_t stream) {
    const int*   z         = (const int*)d_in[0];
    const float* pos       = (const float*)d_in[1];
    const int*   edge_idx  = (const int*)d_in[2];
    const float* edge_mask = (const float*)d_in[3];
    const float* emb       = (const float*)d_in[4];
    const float* mlp_w1    = (const float*)d_in[5];
    const float* mlp_b1    = (const float*)d_in[6];
    const float* mlp_w2    = (const float*)d_in[7];
    const float* mlp_b2    = (const float*)d_in[8];
    const float* lin1_w    = (const float*)d_in[9];
    const float* lin2_w    = (const float*)d_in[10];
    const float* lin2_b    = (const float*)d_in[11];
    const float* lin_w     = (const float*)d_in[12];
    const float* lin_b     = (const float*)d_in[13];

    const int N = in_sizes[0];       // 16384
    const int E = in_sizes[3];       // 600000
    const int* srcIdx = edge_idx;
    const int* dstIdx = edge_idx + E;

    float* h = (float*)d_out;        // N x HID, evolves across layers

    // workspace layout (floats)
    float* ws   = (float*)d_ws;
    float* dArr = ws;                // E
    float* Carr = dArr + E;          // E
    float* hs   = Carr + E;          // N x NF
    float* mArr = hs + (size_t)N * NFF;   // N x NF
    float* xArr = mArr + (size_t)N * NFF; // N x HID

    // setup
    init_h_kernel<<<(N * HIDD + 255) / 256, 256, 0, stream>>>(z, emb, h, N * HIDD);
    edge_geom_kernel<<<(E + 255) / 256, 256, 0, stream>>>(pos, srcIdx, dstIdx,
                                                          edge_mask, dArr, Carr, E);

    int nTiles = E / 16;                       // 37500
    int eBlocks = (nTiles + EWAVES - 1) / EWAVES;

    for (int l = 0; l < NLL; ++l) {
        // hs = h @ lin1_w[l]                 (N x HID) @ (HID x NF)
        launch_gemm(0, h, lin1_w + (size_t)l * HIDD * NFF, nullptr, hs,
                    N, HIDD, NFF, stream);
        // m = 0
        zero_kernel<<<(N * NFF + 255) / 256, 256, 0, stream>>>(mArr, N * NFF);
        // edge message pass: m += scatter(dst, hs[src] * W(edge))
        edge_kernel<<<eBlocks, EWAVES * 32, 0, stream>>>(
            dArr, Carr, srcIdx, dstIdx,
            mlp_w1 + (size_t)l * NGG * NFF, mlp_b1 + (size_t)l * NFF,
            mlp_w2 + (size_t)l * NFF * NFF, mlp_b2 + (size_t)l * NFF,
            hs, mArr, nTiles);
        // x = ssp(m @ lin2_w[l] + lin2_b[l]) (N x NF) @ (NF x HID)
        launch_gemm(1, mArr, lin2_w + (size_t)l * NFF * HIDD,
                    lin2_b + (size_t)l * HIDD, xArr, N, NFF, HIDD, stream);
        // h = h + x @ lin_w[l] + lin_b[l]    (N x HID) @ (HID x HID)
        launch_gemm(2, xArr, lin_w + (size_t)l * HIDD * HIDD,
                    lin_b + (size_t)l * HIDD, h, N, HIDD, HIDD, stream);
    }
}
